// GraphAttentionLayer_1005022347767
// MI455X (gfx1250) — matrix operand order repaired and checked by reading, not compiled, not run on hardware
//
#include <hip/hip_runtime.h>

typedef __attribute__((ext_vector_type(16))) _Float16 v16h;
typedef __attribute__((ext_vector_type(8)))  float    v8f;

#define B_    8
#define N_    2048
#define FIN   256
#define FOUT  128
#define ALPHA 0.2f
#define LOG2E 1.4426950408889634f

// ---------------- Kernel 0: Wt16[n][k] = (f16) W[k][n] ----------------
__global__ void k_wt(const float* __restrict__ W, _Float16* __restrict__ Wt)
{
    int o = blockIdx.x * blockDim.x + threadIdx.x;   // [0, FOUT*FIN)
    if (o >= FOUT * FIN) return;
    int n = o / FIN;
    int k = o % FIN;
    Wt[o] = (_Float16)W[k * FOUT + n];
}

// ------------- Kernel 1: h = x @ W  (WMMA f16, f32 accum) -------------
// One wave per block; wave computes a 16-row x 128-col strip.
// Result stored transposed as f16: h16t[b][col][i].
__global__ void __launch_bounds__(32) k_gemm1(const float* __restrict__ x,
                                              const _Float16* __restrict__ Wt,
                                              _Float16* __restrict__ h16t)
{
    const int lane  = threadIdx.x & 31;
    const int halfu = lane >> 4;          // which 16-lane half
    const int m     = lane & 15;          // row (A) / col (B) within tile
    const int R0    = blockIdx.x * 16;    // global row base (B*N rows flat)
    const int b     = R0 / N_;
    const int i0    = R0 % N_;
    const int kb    = halfu ? 8 : 0;      // A-operand K base for this half

    const float* xrow = x + (size_t)(R0 + m) * FIN;

    v8f c[8] = {};

    for (int k0 = 0; k0 < FIN; k0 += 32) {
        float4 g0 = *(const float4*)(xrow + k0 + kb);
        float4 g1 = *(const float4*)(xrow + k0 + kb + 4);
        float4 g2 = *(const float4*)(xrow + k0 + kb + 16);
        float4 g3 = *(const float4*)(xrow + k0 + kb + 20);

        v16h Bv[8];
#pragma unroll
        for (int t = 0; t < 8; ++t) {
            const _Float16* wp = Wt + (size_t)(t * 16 + m) * FIN + k0 + (halfu ? 8 : 0);
            typedef __attribute__((ext_vector_type(8))) _Float16 v8h_;
            v8h_ lo = *(const v8h_*)wp, hi = *(const v8h_*)(wp + 16);
            Bv[t] = __builtin_shufflevector(lo, hi, 0,1,2,3,4,5,6,7,8,9,10,11,12,13,14,15);
        }

        // A tile 16x32: half<16 holds K = k0 + {0..7, 16..23}, half>=16: {8..15, 24..31}
        v16h A;
        A[0]=(_Float16)g0.x;  A[1]=(_Float16)g0.y;  A[2]=(_Float16)g0.z;  A[3]=(_Float16)g0.w;
        A[4]=(_Float16)g1.x;  A[5]=(_Float16)g1.y;  A[6]=(_Float16)g1.z;  A[7]=(_Float16)g1.w;
        A[8]=(_Float16)g2.x;  A[9]=(_Float16)g2.y;  A[10]=(_Float16)g2.z; A[11]=(_Float16)g2.w;
        A[12]=(_Float16)g3.x; A[13]=(_Float16)g3.y; A[14]=(_Float16)g3.z; A[15]=(_Float16)g3.w;

#pragma unroll
        for (int t = 0; t < 8; ++t)
            c[t] = __builtin_amdgcn_wmma_f32_16x16x32_f16(
                       false, A, false, Bv[t], (short)0, c[t], false, false);
    }

    // store transposed f16: h16t[b][col][i]
#pragma unroll
    for (int t = 0; t < 8; ++t) {
#pragma unroll
        for (int r = 0; r < 8; ++r) {
            int row = halfu ? (r + 8) : r;     // C layout: vgpr r = row r (lanes 0-15) / r+8
            int col = t * 16 + m;
            h16t[((size_t)(b * FOUT + col)) * N_ + (i0 + row)] = (_Float16)c[t][r];
        }
    }
}

// ---------------- Kernel 2: f1 = h@a1, f2 = h@a2 ----------------
__global__ void k_f12(const _Float16* __restrict__ h16t,
                      const float* __restrict__ a,
                      float* __restrict__ f1, float* __restrict__ f2)
{
    int R = blockIdx.x * blockDim.x + threadIdx.x;
    if (R >= B_ * N_) return;
    int b = R / N_, i = R % N_;
    float s1 = 0.f, s2 = 0.f;
    for (int n = 0; n < FOUT; ++n) {
        float h = (float)h16t[((size_t)(b * FOUT + n)) * N_ + i];  // coalesced over i
        s1 += h * a[n];
        s2 += h * a[FOUT + n];
    }
    f1[R] = s1;
    f2[R] = s2;
}

// ---------------- Kernel 3: per-batch max(f2) ----------------
__global__ void k_maxf2(const float* __restrict__ f2, float* __restrict__ maxf2)
{
    __shared__ float sm[256];
    int b = blockIdx.x;
    float mx = -3.0e38f;
    for (int k = threadIdx.x; k < N_; k += 256)
        mx = fmaxf(mx, f2[b * N_ + k]);
    sm[threadIdx.x] = mx;
    __syncthreads();
    for (int s = 128; s > 0; s >>= 1) {
        if (threadIdx.x < s) sm[threadIdx.x] = fmaxf(sm[threadIdx.x], sm[threadIdx.x + s]);
        __syncthreads();
    }
    if (threadIdx.x == 0) maxf2[b] = sm[0];
}

// ------- Kernel 4: fused masked-softmax attention @ h, ELU (flash-style) -------
// One wave per block, 16 rows per wave, streaming j in tiles of 32.
// adj stream is software-pipelined one tile ahead. All softmax logit math is done
// in the log2-scaled domain (valid: leaky_relu is positively homogeneous), so the
// per-element chain is: add, mul+max (leaky), sub, cndmask (mask), v_exp, cvt.
__global__ void __launch_bounds__(32) k_attn(const int* __restrict__ adj,
                                             const _Float16* __restrict__ h16t,
                                             const float* __restrict__ f1,
                                             const float* __restrict__ f2,
                                             const float* __restrict__ maxf2,
                                             float* __restrict__ out)
{
    __shared__ float f2s[N_];                  // f2 * log2(e)
    const int lane  = threadIdx.x & 31;
    const int halfu = lane >> 4;
    const int m     = lane & 15;
    const int kb    = halfu ? 8 : 0;

    const int tile = blockIdx.x;               // [0, B*N/16)
    const int b    = tile / (N_ / 16);
    const int i0   = (tile % (N_ / 16)) * 16;

    for (int k = lane; k < N_; k += 32) f2s[k] = f2[b * N_ + k] * LOG2E;
    __syncthreads();

    const float f1r = f1[b * N_ + i0 + m];
    const float mf2 = maxf2[b];
    float sv  = f1r + mf2;
    sv        = fmaxf(sv, ALPHA * sv);         // leaky: upper bound on this row's max logit
    const float svl = sv  * LOG2E;             // shift, log2 domain
    const float f1l = f1r * LOG2E;             // row logit part, log2 domain

    const int* adjRow = adj + ((size_t)(b * N_ + i0 + m)) * N_;

    v8f   c[8] = {};
    float psum = 0.f;

    // prime the adj pipeline (tile j0 = 0)
    int4 q0 = *(const int4*)(adjRow + kb);
    int4 q1 = *(const int4*)(adjRow + kb + 4);
    int4 q2 = *(const int4*)(adjRow + kb + 16);
    int4 q3 = *(const int4*)(adjRow + kb + 20);

    for (int j0 = 0; j0 < N_; j0 += 32) {
        // stage next adj tile (wrap-masked: uniform control flow, EXEC stays all-1s)
        const int jn = (j0 + 32) & (N_ - 1);
        int4 n0 = *(const int4*)(adjRow + jn + kb);
        int4 n1 = *(const int4*)(adjRow + jn + kb + 4);
        int4 n2 = *(const int4*)(adjRow + jn + kb + 16);
        int4 n3 = *(const int4*)(adjRow + jn + kb + 20);
        // prefetch adj further ahead into cache (global_prefetch_b8, no LOADcnt)
        __builtin_prefetch(adjRow + ((j0 + 512) & (N_ - 1)), 0, 0);

        // B tiles (two 16B pieces per lane, starting 32B apart); scheduler places these as it sees fit
        v16h Bv[8];
#pragma unroll
        for (int t = 0; t < 8; ++t) {
            const _Float16* hp = h16t + ((size_t)(b * FOUT + t * 16 + m)) * N_
                                      + j0 + (halfu ? 8 : 0);
            typedef __attribute__((ext_vector_type(8))) _Float16 v8h_;
            v8h_ lo = *(const v8h_*)hp, hi = *(const v8h_*)(hp + 16);
            Bv[t] = __builtin_shufflevector(lo, hi, 0,1,2,3,4,5,6,7,8,9,10,11,12,13,14,15);
        }

        // build P tile (A operand) from current adj tile, log2 domain
        int av[16] = { q0.x, q0.y, q0.z, q0.w,  q1.x, q1.y, q1.z, q1.w,
                       q2.x, q2.y, q2.z, q2.w,  q3.x, q3.y, q3.z, q3.w };
        v16h A;
#pragma unroll
        for (int e2 = 0; e2 < 16; ++e2) {
            int K = kb + (e2 < 8 ? e2 : e2 + 8);
            float el  = f1l + f2s[j0 + K];            // logit * log2e
            float lk  = fmaxf(el, ALPHA * el);        // leaky_relu (homogeneous)
            float arg = lk - svl;                     // <= 0
            arg = (av[e2] > 0) ? arg : -200.0f;       // mask: exp2(-200) == 0.0f
            float p = exp2f(arg);
            psum += p;
            A[e2] = (_Float16)p;
        }

        // 8 independent WMMAs back-to-back (XDL, co-executes with next tile's VALU)
#pragma unroll
        for (int t = 0; t < 8; ++t)
            c[t] = __builtin_amdgcn_wmma_f32_16x16x32_f16(
                       false, A, false, Bv[t], (short)0, c[t], false, false);

        q0 = n0; q1 = n1; q2 = n2; q3 = n3;
    }

    // complete per-row denominators: lane m + lane m+16 hold complementary K halves
    float lfull = psum + __shfl_xor(psum, 16, 32);

#pragma unroll
    for (int r = 0; r < 8; ++r) {
        int   row = halfu ? (r + 8) : r;
        float lr  = __shfl(lfull, row, 32);    // row sums live one-row-per-lane (0..15)
        float inv = 1.0f / lr;
#pragma unroll
        for (int t = 0; t < 8; ++t) {
            float v = c[t][r] * inv;
            v = v > 0.f ? v : (expf(v) - 1.0f);                    // ELU (alpha=1)
            out[((size_t)(b * N_ + i0 + row)) * FOUT + t * 16 + m] = v;
        }
    }
}

extern "C" void kernel_launch(void* const* d_in, const int* in_sizes, int n_in,
                              void* d_out, int out_size, void* d_ws, size_t ws_size,
                              hipStream_t stream)
{
    const float* x   = (const float*)d_in[0];
    const int*   adj = (const int*)d_in[1];
    const float* W   = (const float*)d_in[2];
    const float* a   = (const float*)d_in[3];
    float* out = (float*)d_out;

    char* ws = (char*)d_ws;
    _Float16* Wt   = (_Float16*)(ws);                               // 64 KB
    _Float16* h16t = (_Float16*)(ws + (64 << 10));                  // 4 MB
    float*    f1   = (float*)(ws + (64 << 10) + (4 << 20));         // 64 KB
    float*    f2   = (float*)(ws + (128 << 10) + (4 << 20));        // 64 KB
    float*    mf2  = (float*)(ws + (192 << 10) + (4 << 20));        // 32 B

    hipLaunchKernelGGL(k_wt,    dim3((FOUT * FIN + 255) / 256), dim3(256), 0, stream, W, Wt);
    hipLaunchKernelGGL(k_gemm1, dim3(B_ * N_ / 16), dim3(32), 0, stream, x, Wt, h16t);
    hipLaunchKernelGGL(k_f12,   dim3((B_ * N_ + 255) / 256), dim3(256), 0, stream, h16t, a, f1, f2);
    hipLaunchKernelGGL(k_maxf2, dim3(B_), dim3(256), 0, stream, f2, mf2);
    hipLaunchKernelGGL(k_attn,  dim3(B_ * N_ / 16), dim3(32), 0, stream,
                       adj, h16t, f1, f2, mf2, out);
}
